// Attention_3049426780849
// MI455X (gfx1250) — compile-verified, matrix operand order checked by reading, not run on hardware
//
#include <hip/hip_runtime.h>

typedef __attribute__((ext_vector_type(16))) __bf16 v16bf;
typedef __attribute__((ext_vector_type(8)))  float  v8f;
typedef __attribute__((ext_vector_type(4)))  unsigned int u32x4;
typedef __attribute__((ext_vector_type(8)))  unsigned int u32x8;

#define DIM   1024
#define HEADS 16
#define DHEAD 64
#define NSEQ  2048
#define BATCH 4
#define ROWS  (BATCH * NSEQ)   // 8192

__device__ __forceinline__ unsigned short f2bf(float f) {
    union { float f; unsigned int u; } a; a.f = f;
    unsigned int u = a.u;
    u += 0x7FFFu + ((u >> 16) & 1u);   // round-to-nearest-even
    return (unsigned short)(u >> 16);
}

__device__ __forceinline__ unsigned lds_off(const void* p) {
    // generic shared pointer: addr[31:0] is the LDS byte address (aperture in high bits)
    return (unsigned)(size_t)p;
}

// ---------------------------------------------------------------- TDM -------
// Issue a 2D tensor_load_to_lds (CDNA5 ISA ch.8): bf16 elements, tile
// (tile_w x tile_h) from a row-major tensor with row stride `stride_elems`,
// written to LDS with hardware padding so LDS row stride =
// tile_w + 4*(pad_amount_code+1)/... (pad codes chosen by caller).
__device__ __forceinline__ void tdm_load_2d(const unsigned short* gptr,
                                            unsigned lds_byte_addr,
                                            unsigned tile_w, unsigned tile_h,
                                            unsigned stride_elems,
                                            unsigned pad_interval_code,
                                            unsigned pad_amount_code) {
    unsigned long long ga = (unsigned long long)(size_t)gptr;
    u32x4 g0;
    g0[0] = 1u;                                    // count=1, user descriptor
    g0[1] = lds_byte_addr;                         // lds_addr
    g0[2] = (unsigned)ga;                          // global_addr[31:0]
    g0[3] = ((unsigned)(ga >> 32) & 0x01FFFFFFu)   // global_addr[56:32]
          | 0x80000000u;                           // type=2 ("image")
    u32x8 g1;
    g1[0] = (1u << 16)                             // data_size = 1 -> 2 bytes
          | (1u << 20)                             // pad_enable
          | (pad_interval_code << 22)
          | (pad_amount_code << 25);
    g1[1] = (stride_elems & 0xFFFFu) << 16;        // tensor_dim0[15:0]
    g1[2] = (stride_elems >> 16);                  // tensor_dim0[31:16], dim1 lo16 = 0
    g1[3] = 16u                                    // tensor_dim1 = 0x100000 rows
          | ((tile_w & 0xFFFFu) << 16);            // tile_dim0
    g1[4] = tile_h & 0xFFFFu;                      // tile_dim1 (tile_dim2 = 0)
    g1[5] = stride_elems;                          // tensor_dim0_stride[31:0]
    g1[6] = 0u;                                    // stride[47:32], dim1_stride lo
    g1[7] = 0u;
    asm volatile("tensor_load_to_lds %0, %1" :: "s"(g0), "s"(g1) : "memory");
}

// Load a 16x32 bf16 A/B fragment from LDS per CDNA5 WMMA layout:
// lane%16 = row (A: M, B: N), lane/16 selects K-half; per lane two
// contiguous 8-element runs: [kOff + half*8, +8) and [kOff + 16 + half*8, +8).
__device__ __forceinline__ v16bf frag_ld(const unsigned short* s, int stride,
                                         int row, int kOff, int khalf) {
    union { v16bf v; uint4 q[2]; } u;
    const unsigned short* base = s + row * stride + kOff + khalf * 8;
    u.q[0] = *(const uint4*)(base);
    u.q[1] = *(const uint4*)(base + 16);
    return u.v;
}

__device__ __forceinline__ v8f wmma_bf16(v16bf a, v16bf b, v8f c) {
    return __builtin_amdgcn_wmma_f32_16x16x32_bf16(false, a, false, b,
                                                   (short)0, c, false, false);
}

// ---------------------------------------------------------------- cast+T ----
// out[n*K + k] = bf16(in[k*N + n])   (weights: K-major fp32 -> N-major bf16)
__global__ __launch_bounds__(256) void castT_kernel(const float* __restrict__ in,
                                                    unsigned short* __restrict__ out,
                                                    int K, int N) {
    size_t idx = (size_t)blockIdx.x * 256 + threadIdx.x;
    if (idx >= (size_t)K * N) return;
    int n = (int)(idx / K);
    int k = (int)(idx % K);
    out[idx] = f2bf(in[(size_t)k * N + n]);
}

// ------------------------------------------------------------- layernorm ----
__global__ __launch_bounds__(256) void ln_kernel(const float* __restrict__ x,
                                                 const float* __restrict__ gamma,
                                                 const float* __restrict__ beta,
                                                 unsigned short* __restrict__ xn) {
    const int row  = blockIdx.x;
    const int tid  = threadIdx.x;
    const int wave = tid >> 5, lane = tid & 31;
    const float* xr = x + (size_t)row * DIM;

    float v[4], s = 0.f, s2 = 0.f;
#pragma unroll
    for (int i = 0; i < 4; i++) {
        v[i] = xr[tid + i * 256];
        s  += v[i];
        s2 += v[i] * v[i];
    }
#pragma unroll
    for (int m = 16; m >= 1; m >>= 1) {
        s  += __shfl_xor(s,  m, 32);
        s2 += __shfl_xor(s2, m, 32);
    }
    __shared__ float red[16];
    if (lane == 0) { red[wave] = s; red[wave + 8] = s2; }
    __syncthreads();
    s = 0.f; s2 = 0.f;
#pragma unroll
    for (int w = 0; w < 8; w++) { s += red[w]; s2 += red[w + 8]; }
    float mu  = s * (1.0f / DIM);
    float var = s2 * (1.0f / DIM) - mu * mu;
    float inv = rsqrtf(var + 1e-5f);
#pragma unroll
    for (int i = 0; i < 4; i++) {
        int c = tid + i * 256;
        xn[(size_t)row * DIM + c] = f2bf((v[i] - mu) * inv * gamma[c] + beta[c]);
    }
}

// ------------------------------------------------------------- QKV GEMM -----
// xn [8192,1024] bf16 @ wqkv_t [3072(N),1024(K)] bf16 -> scatter q,k,v(T)
#define GSTRIDE 40   // 32 K + 8 pad ushorts (80B rows; TDM pad: 64B rows + 16B)

__global__ __launch_bounds__(256) void qkv_gemm_kernel(
        const unsigned short* __restrict__ xn,
        const unsigned short* __restrict__ wt,
        unsigned short* __restrict__ q,     // [b,h,n,d]
        unsigned short* __restrict__ k,     // [b,h,n,d]
        unsigned short* __restrict__ v) {   // [b,h,d,n]  (pre-transposed)
    __shared__ __align__(16) unsigned short As[2][128 * GSTRIDE];
    __shared__ __align__(16) unsigned short Bs[2][128 * GSTRIDE];

    const int tid  = threadIdx.x;
    const int wave = tid >> 5, lane = tid & 31;
    const int waveM = wave >> 1, waveN = wave & 1;
    const int hi = lane >> 4, lo = lane & 15;
    const int bm = blockIdx.y * 128;
    const int bn = blockIdx.x * 128;

    const unsigned short* aTile = xn + (size_t)bm * DIM;
    const unsigned short* bTile = wt + (size_t)bn * DIM;

    v8f acc[2][4];
#pragma unroll
    for (int i = 0; i < 2; i++)
#pragma unroll
        for (int j = 0; j < 4; j++)
#pragma unroll
            for (int e = 0; e < 8; e++) acc[i][j][e] = 0.0f;

    if (wave == 0) {   // prologue: DMA first tile pair into buffer 0
        tdm_load_2d(aTile, lds_off(As[0]), 32, 128, DIM, 3, 3);
        tdm_load_2d(bTile, lds_off(Bs[0]), 32, 128, DIM, 3, 3);
    }

    int cur = 0;
    for (int kt = 0; kt < DIM; kt += 32) {
        if (wave == 0) {
            if (kt + 32 < DIM) {   // prefetch next pair, overlap with compute
                tdm_load_2d(aTile + kt + 32, lds_off(As[cur ^ 1]), 32, 128, DIM, 3, 3);
                tdm_load_2d(bTile + kt + 32, lds_off(Bs[cur ^ 1]), 32, 128, DIM, 3, 3);
                __builtin_amdgcn_s_wait_tensorcnt(2);  // current pair done
            } else {
                __builtin_amdgcn_s_wait_tensorcnt(0);
            }
        }
        __syncthreads();

        v16bf af[2], bf[4];
#pragma unroll
        for (int mt = 0; mt < 2; mt++)
            af[mt] = frag_ld(As[cur], GSTRIDE, waveM * 32 + mt * 16 + lo, 0, hi);
#pragma unroll
        for (int nt = 0; nt < 4; nt++)
            bf[nt] = frag_ld(Bs[cur], GSTRIDE, waveN * 64 + nt * 16 + lo, 0, hi);
#pragma unroll
        for (int mt = 0; mt < 2; mt++)
#pragma unroll
            for (int nt = 0; nt < 4; nt++)
                acc[mt][nt] = wmma_bf16(af[mt], bf[nt], acc[mt][nt]);
        __syncthreads();
        cur ^= 1;
    }

    const int sec = bn >> 10;                       // 0:q 1:k 2:v (128 | 1024)
    unsigned short* dst = (sec == 0) ? q : (sec == 1) ? k : v;
    const float scale = (sec == 0) ? 0.125f : 1.0f; // DHEAD^-0.5 folded into Q
#pragma unroll
    for (int mt = 0; mt < 2; mt++) {
        int gRowBase = bm + waveM * 32 + mt * 16 + hi * 8;
#pragma unroll
        for (int nt = 0; nt < 4; nt++) {
            int gCol = bn + waveN * 64 + nt * 16 + lo;
            int within = gCol & (DIM - 1);
            int h = within >> 6, d = within & 63;
#pragma unroll
            for (int r = 0; r < 8; r++) {
                int gRow = gRowBase + r;
                int bidx = gRow >> 11, nn = gRow & (NSEQ - 1);
                size_t o = (sec == 2)
                    ? ((size_t)(bidx * HEADS + h) * DHEAD + d) * NSEQ + nn
                    : ((size_t)(bidx * HEADS + h) * NSEQ + nn) * DHEAD + d;
                dst[o] = f2bf(acc[mt][nt][r] * scale);
            }
        }
    }
}

// ------------------------------------------------------ flash attention -----
#define QSTRIDE 72   // 64 + 8 pad ushorts (144B rows; TDM pad: 128B rows + 16B)

__global__ __launch_bounds__(256) void attn_kernel(
        const unsigned short* __restrict__ qg,    // [b,h,n,d]
        const unsigned short* __restrict__ kg,    // [b,h,n,d]
        const unsigned short* __restrict__ vtg,   // [b,h,d,n]
        const float* __restrict__ bias,
        unsigned short* __restrict__ attn_out) {
    __shared__ __align__(16) unsigned short Qs[128 * QSTRIDE];
    __shared__ __align__(16) unsigned short Ks[64 * QSTRIDE];    // [key][d]
    __shared__ __align__(16) unsigned short Vts[64 * QSTRIDE];   // [d][key]
    __shared__ __align__(16) unsigned short Ps[8 * 16 * QSTRIDE];

    const int tid  = threadIdx.x;
    const int wave = tid >> 5, lane = tid & 31;
    const int hi = lane >> 4, lo = lane & 15;
    const int bh = blockIdx.y;                 // bidx*16 + h
    const int bidx = bh >> 4, h = bh & 15;
    const int qBase = blockIdx.x * 128;
    const size_t headBase = (size_t)bh * NSEQ * DHEAD;

    if (wave == 0) {   // Q tile 128x64 via TDM
        tdm_load_2d(qg + headBase + (size_t)qBase * DHEAD, lds_off(Qs),
                    64, 128, DHEAD, 4, 3);
        __builtin_amdgcn_s_wait_tensorcnt(0);
    }
    __syncthreads();

    v16bf qfrag[2];
#pragma unroll
    for (int ks = 0; ks < 2; ks++)
        qfrag[ks] = frag_ld(Qs, QSTRIDE, wave * 16 + lo, ks * 32, hi);

    float m[8], l[8];
    v8f oAcc[4];
#pragma unroll
    for (int r = 0; r < 8; r++) { m[r] = -1e30f; l[r] = 0.f; }
#pragma unroll
    for (int dt = 0; dt < 4; dt++)
#pragma unroll
        for (int e = 0; e < 8; e++) oAcc[dt][e] = 0.f;

    unsigned short* Pw = Ps + wave * 16 * QSTRIDE;

    for (int kb = 0; kb < NSEQ; kb += 64) {
        __syncthreads();   // everyone done reading previous K/V tiles
        if (wave == 0) {
            tdm_load_2d(kg + headBase + (size_t)kb * DHEAD, lds_off(Ks),
                        64, 64, DHEAD, 4, 3);
            tdm_load_2d(vtg + headBase + kb, lds_off(Vts),
                        64, 64, NSEQ, 4, 3);
            __builtin_amdgcn_s_wait_tensorcnt(0);
        }
        __syncthreads();

        // S = Q K^T  (K's [n][d] layout == WMMA B-fragment layout)
        v8f sAcc[4];
#pragma unroll
        for (int nt = 0; nt < 4; nt++) {
#pragma unroll
            for (int e = 0; e < 8; e++) sAcc[nt][e] = 0.f;
#pragma unroll
            for (int ks = 0; ks < 2; ks++) {
                v16bf kf = frag_ld(Ks, QSTRIDE, nt * 16 + lo, ks * 32, hi);
                sAcc[nt] = wmma_bf16(qfrag[ks], kf, sAcc[nt]);
            }
        }
        // + relative position bias (256 MB stream -> NT loads, the HBM floor)
        const float* bb = bias + ((size_t)h * NSEQ + qBase + wave * 16 + hi * 8) * NSEQ
                               + kb + lo;
#pragma unroll
        for (int nt = 0; nt < 4; nt++)
#pragma unroll
            for (int r = 0; r < 8; r++)
                sAcc[nt][r] += __builtin_nontemporal_load(bb + (size_t)r * NSEQ + nt * 16);

        // row max across 4 tiles x 16 lanes of the half-wave
        float corr[8], rs[8];
#pragma unroll
        for (int r = 0; r < 8; r++) {
            float mx = fmaxf(fmaxf(sAcc[0][r], sAcc[1][r]),
                             fmaxf(sAcc[2][r], sAcc[3][r]));
#pragma unroll
            for (int msk = 8; msk >= 1; msk >>= 1)
                mx = fmaxf(mx, __shfl_xor(mx, msk, 32));
            float mNew = fmaxf(m[r], mx);
            corr[r] = __expf(m[r] - mNew);
            m[r] = mNew;
            rs[r] = 0.f;
        }
        // P = exp(S - m), stage bf16 in per-wave LDS strip
#pragma unroll
        for (int nt = 0; nt < 4; nt++)
#pragma unroll
            for (int r = 0; r < 8; r++) {
                float p = __expf(sAcc[nt][r] - m[r]);
                rs[r] += p;
                Pw[(r + hi * 8) * QSTRIDE + nt * 16 + lo] = f2bf(p);
            }
#pragma unroll
        for (int r = 0; r < 8; r++) {
#pragma unroll
            for (int msk = 8; msk >= 1; msk >>= 1)
                rs[r] += __shfl_xor(rs[r], msk, 32);
            l[r] = l[r] * corr[r] + rs[r];
        }
#pragma unroll
        for (int dt = 0; dt < 4; dt++)
#pragma unroll
            for (int r = 0; r < 8; r++) oAcc[dt][r] *= corr[r];

        // O += P @ V   (B fragment direct from TDM-loaded V^T tile)
        v16bf pf[2];
#pragma unroll
        for (int ks2 = 0; ks2 < 2; ks2++)
            pf[ks2] = frag_ld(Pw, QSTRIDE, lo, ks2 * 32, hi);
#pragma unroll
        for (int dt = 0; dt < 4; dt++)
#pragma unroll
            for (int ks2 = 0; ks2 < 2; ks2++) {
                v16bf vtf = frag_ld(Vts, QSTRIDE, dt * 16 + lo, ks2 * 32, hi);
                oAcc[dt] = wmma_bf16(pf[ks2], vtf, oAcc[dt]);
            }
    }

    // finalize: O / l  -> attn_out [b, n, 1024] bf16
#pragma unroll
    for (int dt = 0; dt < 4; dt++)
#pragma unroll
        for (int r = 0; r < 8; r++) {
            int qrow = qBase + wave * 16 + hi * 8 + r;
            float val = oAcc[dt][r] / l[r];
            attn_out[((size_t)(bidx * NSEQ + qrow)) * DIM + h * DHEAD + dt * 16 + lo]
                = f2bf(val);
        }
}

// ------------------------------------------------------------- out GEMM -----
// attn [8192,1024] bf16 @ wout_t [1024(N),1024(K)] bf16 + b_out -> fp32 out
__global__ __launch_bounds__(256) void out_gemm_kernel(
        const unsigned short* __restrict__ a,
        const unsigned short* __restrict__ wt,
        const float* __restrict__ b_out,
        float* __restrict__ out) {
    __shared__ __align__(16) unsigned short As[2][128 * GSTRIDE];
    __shared__ __align__(16) unsigned short Bs[2][128 * GSTRIDE];

    const int tid  = threadIdx.x;
    const int wave = tid >> 5, lane = tid & 31;
    const int waveM = wave >> 1, waveN = wave & 1;
    const int hi = lane >> 4, lo = lane & 15;
    const int bm = blockIdx.y * 128;
    const int bn = blockIdx.x * 128;

    const unsigned short* aTile = a  + (size_t)bm * DIM;
    const unsigned short* bTile = wt + (size_t)bn * DIM;

    v8f acc[2][4];
#pragma unroll
    for (int i = 0; i < 2; i++)
#pragma unroll
        for (int j = 0; j < 4; j++)
#pragma unroll
            for (int e = 0; e < 8; e++) acc[i][j][e] = 0.0f;

    if (wave == 0) {
        tdm_load_2d(aTile, lds_off(As[0]), 32, 128, DIM, 3, 3);
        tdm_load_2d(bTile, lds_off(Bs[0]), 32, 128, DIM, 3, 3);
    }

    int cur = 0;
    for (int kt = 0; kt < DIM; kt += 32) {
        if (wave == 0) {
            if (kt + 32 < DIM) {
                tdm_load_2d(aTile + kt + 32, lds_off(As[cur ^ 1]), 32, 128, DIM, 3, 3);
                tdm_load_2d(bTile + kt + 32, lds_off(Bs[cur ^ 1]), 32, 128, DIM, 3, 3);
                __builtin_amdgcn_s_wait_tensorcnt(2);
            } else {
                __builtin_amdgcn_s_wait_tensorcnt(0);
            }
        }
        __syncthreads();

        v16bf af[2], bf[4];
#pragma unroll
        for (int mt = 0; mt < 2; mt++)
            af[mt] = frag_ld(As[cur], GSTRIDE, waveM * 32 + mt * 16 + lo, 0, hi);
#pragma unroll
        for (int nt = 0; nt < 4; nt++)
            bf[nt] = frag_ld(Bs[cur], GSTRIDE, waveN * 64 + nt * 16 + lo, 0, hi);
#pragma unroll
        for (int mt = 0; mt < 2; mt++)
#pragma unroll
            for (int nt = 0; nt < 4; nt++)
                acc[mt][nt] = wmma_bf16(af[mt], bf[nt], acc[mt][nt]);
        __syncthreads();
        cur ^= 1;
    }

#pragma unroll
    for (int mt = 0; mt < 2; mt++) {
        int gRowBase = bm + waveM * 32 + mt * 16 + hi * 8;
#pragma unroll
        for (int nt = 0; nt < 4; nt++) {
            int gCol = bn + waveN * 64 + nt * 16 + lo;
            float bv = b_out[gCol];
#pragma unroll
            for (int r = 0; r < 8; r++)
                out[(size_t)(gRowBase + r) * DIM + gCol] = acc[mt][nt][r] + bv;
        }
    }
}

// --------------------------------------------------------------- launch -----
extern "C" void kernel_launch(void* const* d_in, const int* in_sizes, int n_in,
                              void* d_out, int out_size, void* d_ws, size_t ws_size,
                              hipStream_t stream) {
    const float* x      = (const float*)d_in[0];
    const float* bias   = (const float*)d_in[1];
    const float* w_qkv  = (const float*)d_in[2];
    const float* w_out  = (const float*)d_in[3];
    const float* b_out  = (const float*)d_in[4];
    const float* gamma  = (const float*)d_in[5];
    const float* beta   = (const float*)d_in[6];
    float* out = (float*)d_out;

    char* ws = (char*)d_ws;
    unsigned short* xn    = (unsigned short*)(ws);              // 16 MB
    unsigned short* wqkvt = (unsigned short*)(ws + 16777216);   //  6 MB
    unsigned short* woutt = (unsigned short*)(ws + 23068672);   //  2 MB
    unsigned short* qb    = (unsigned short*)(ws + 25165824);   // 16 MB [b,h,n,d]
    unsigned short* kb    = (unsigned short*)(ws + 41943040);   // 16 MB [b,h,n,d]
    unsigned short* vtb   = (unsigned short*)(ws + 58720256);   // 16 MB [b,h,d,n]
    unsigned short* attn  = (unsigned short*)(ws + 75497472);   // 16 MB

    castT_kernel<<<(3072 * 1024) / 256, 256, 0, stream>>>(w_qkv, wqkvt, 1024, 3072);
    castT_kernel<<<(1024 * 1024) / 256, 256, 0, stream>>>(w_out, woutt, 1024, 1024);
    ln_kernel<<<ROWS, 256, 0, stream>>>(x, gamma, beta, xn);
    qkv_gemm_kernel<<<dim3(24, 64), 256, 0, stream>>>(xn, wqkvt, qb, kb, vtb);
    attn_kernel<<<dim3(16, 64), 256, 0, stream>>>(qb, kb, vtb, bias, attn);
    out_gemm_kernel<<<dim3(8, 64), 256, 0, stream>>>(attn, woutt, b_out, out);
}
